// SlabCrossAttention3D2D_25778393710977
// MI455X (gfx1250) — compile-verified, matrix operand order checked by reading, not run on hardware
//
#include <hip/hip_runtime.h>
#include <hip/hip_bf16.h>

typedef _Float16 h16;
typedef __attribute__((ext_vector_type(16))) _Float16 v16h;
typedef __attribute__((ext_vector_type(8)))  _Float16 v8h;
typedef __attribute__((ext_vector_type(8)))  float    v8f;

#define COMB16(lo, hi) __builtin_shufflevector((lo),(hi),0,1,2,3,4,5,6,7,8,9,10,11,12,13,14,15)

// ---- problem constants -----------------------------------------------------
#define NB   2
#define C3_  128
#define DZ   32
#define HH_  24
#define WW_  24
#define HW_  576
#define C2_  96
#define KK_  8
#define EE   256
#define NHD  4
#define HD_  64
#define THID 64
#define EPSF 1e-5f

// ---- workspace layout (bytes, all 256-aligned) -----------------------------
static constexpr size_t WS_XQ   = 0;
static constexpr size_t WS_XKV  = WS_XQ   + (size_t)36864*128*2;
static constexpr size_t WS_WQT  = WS_XKV  + (size_t)9216*96*2;
static constexpr size_t WS_WKT  = WS_WQT  + (size_t)256*128*2;
static constexpr size_t WS_WVT  = WS_WKT  + (size_t)256*96*2;
static constexpr size_t WS_WPT  = WS_WVT  + (size_t)256*96*2;
static constexpr size_t WS_Q    = WS_WPT  + (size_t)128*256*2;
static constexpr size_t WS_K    = WS_Q    + (size_t)36864*256*2;
static constexpr size_t WS_V    = WS_K    + (size_t)9216*256*2;
static constexpr size_t WS_O    = WS_V    + (size_t)9216*256*2;
static constexpr size_t WS_H    = WS_O    + (size_t)36864*256*2;
static constexpr size_t WS_P    = WS_H    + (size_t)589824*4;
static constexpr size_t WS_T    = WS_P    + (size_t)9216*4;

// ---- small helpers ---------------------------------------------------------
__device__ __forceinline__ v8f wmma_f16(v16h a, v16h b, v8f c) {
  return __builtin_amdgcn_wmma_f32_16x16x32_f16(false, a, false, b, (short)0, c, false, false);
}
// 32-wide (k) fragment slice for this lane from a 16B-aligned k-contiguous row:
// halves {grp*8..+7} and {16+grp*8..+7} (documented 16-bit A layout; mirrored B)
__device__ __forceinline__ v16h frag32(const h16* rowbase, int grp) {
  v8h lo = *(const v8h*)(rowbase + grp*8);
  v8h hi = *(const v8h*)(rowbase + 16 + grp*8);
  return COMB16(lo, hi);
}
template<int PAT>
__device__ __forceinline__ float swz(float x) {
  return __int_as_float(__builtin_amdgcn_ds_swizzle(__float_as_int(x), PAT));
}
__device__ __forceinline__ float redmax16(float x) {   // reduce across 16-lane group
  x = fmaxf(x, swz<0x041F>(x)); x = fmaxf(x, swz<0x081F>(x));
  x = fmaxf(x, swz<0x101F>(x)); x = fmaxf(x, swz<0x201F>(x));
  return x;
}
__device__ __forceinline__ float redsum16(float x) {
  x += swz<0x041F>(x); x += swz<0x081F>(x);
  x += swz<0x101F>(x); x += swz<0x201F>(x);
  return x;
}

// ---- stage 0: conversions & packing ---------------------------------------
// dst[n*rows + k] = (f16) src[k*cols + n]   (W is [k=rows][n=cols] fp32)
__global__ void cvt_transpose_f16(const float* __restrict__ s, h16* __restrict__ d,
                                  int rows, int cols) {
  int i = blockIdx.x*256 + threadIdx.x;
  if (i >= rows*cols) return;
  int n = i / rows, k = i % rows;
  d[i] = (h16)s[(size_t)k*cols + n];
}
__global__ void pack_xq(const float* __restrict__ F3, h16* __restrict__ X) {
  int i = blockIdx.x*256 + threadIdx.x;            // 36864*128
  if (i >= 36864*128) return;
  int c = i & 127, row = i >> 7;
  int hw = row % HW_, bz = row / HW_, z = bz % DZ, b = bz / DZ;
  X[i] = (h16)F3[((size_t)(b*C3_ + c)*DZ + z)*HW_ + hw];
}
__global__ void pack_xkv(const float* __restrict__ F2, h16* __restrict__ X) {
  int i = blockIdx.x*256 + threadIdx.x;            // 9216*96
  if (i >= 9216*96) return;
  int c = i % C2_, row = i / C2_;
  int yx = row % HW_, bk = row / HW_, kk = bk & 7, b = bk >> 3;
  X[i] = (h16)F2[(((size_t)(b*C2_ + c)*HW_) + yx)*KK_ + kk];
}

// ---- stage 1: topo branch (scalar, ~1% of FLOPs) ---------------------------
__global__ void topo_conv1(const float* __restrict__ F2, const float* __restrict__ w,
                           const float* __restrict__ bn_g, const float* __restrict__ bn_b,
                           float* __restrict__ hbuf) {
  int i = blockIdx.x*256 + threadIdx.x;            // 16*64*576
  if (i >= 589824) return;
  int yx = i % HW_, th = (i / HW_) % THID, bk = i / (HW_*THID);
  int y = yx / WW_, x = yx % WW_, kk = bk & 7, b = bk >> 3;
  float acc = 0.f;
  for (int c = 0; c < C2_; ++c) {
    const float* f2c = F2 + ((size_t)(b*C2_ + c)*HW_)*KK_ + kk;
    const float* wc  = w + (th*C2_ + c)*9;
    #pragma unroll
    for (int dy = -1; dy <= 1; ++dy) {
      int yy = y + dy; if (yy < 0 || yy >= HH_) continue;
      #pragma unroll
      for (int dx = -1; dx <= 1; ++dx) {
        int xx = x + dx; if (xx < 0 || xx >= WW_) continue;
        acc += f2c[(size_t)(yy*WW_ + xx)*KK_] * wc[(dy+1)*3 + (dx+1)];
      }
    }
  }
  float v = acc * (bn_g[th] * rsqrtf(1.f + EPSF)) + bn_b[th];
  hbuf[i] = fmaxf(v, 0.f);
}
__global__ void topo_conv2(const float* __restrict__ hbuf, const float* __restrict__ w2,
                           const float* __restrict__ b2, float* __restrict__ pbuf) {
  int i = blockIdx.x*256 + threadIdx.x;            // 9216
  if (i >= 9216) return;
  int yx = i % HW_, bk = i / HW_;
  float acc = b2[0];
  for (int th = 0; th < THID; ++th) acc += hbuf[(size_t)(bk*THID + th)*HW_ + yx] * w2[th];
  pbuf[i] = acc;
}
__global__ void topo_nei(const float* __restrict__ pbuf, float* __restrict__ topo) {
  int i = blockIdx.x*256 + threadIdx.x;            // 9216
  if (i >= 9216) return;
  int yx = i % HW_, bk = i / HW_, y = yx / WW_, x = yx % WW_;
  float s = 0.f;
  #pragma unroll
  for (int dy = -1; dy <= 1; ++dy) {
    int yy = y + dy; if (yy < 0 || yy >= HH_) continue;
    #pragma unroll
    for (int dx = -1; dx <= 1; ++dx) {
      if (dy == 0 && dx == 0) continue;
      int xx = x + dx; if (xx < 0 || xx >= WW_) continue;
      s += pbuf[bk*HW_ + yy*WW_ + xx];
    }
  }
  topo[i] = pbuf[i] + 0.5f * s;
}

// ---- stage 2: WMMA GEMM (X[M,Cin] @ W[Cin,256]) + fused LayerNorm ----------
// A and B fragments both load directly from global (k-contiguous b128 pairs);
// LDS used only for the cross-wave LayerNorm exchange.
__global__ __launch_bounds__(256)
void gemm_ln(const h16* __restrict__ X, const h16* __restrict__ WT,
             const float* __restrict__ gamma, const float* __restrict__ beta,
             h16* __restrict__ Y, int Cin) {
  __shared__ float ldsAcc[16*256];
  __shared__ float s_mean[16], s_rstd[16];
  int t = threadIdx.x, w = t >> 5, lane = t & 31;
  int n16 = lane & 15, grp = lane >> 4;
  int rowbase = blockIdx.x * 16;
  const h16* xa  = X  + (size_t)(rowbase + n16)*Cin;     // A row for this lane
  const h16* wb0 = WT + (size_t)(w*32 + n16)*Cin;        // B column, nt=0
  const h16* wb1 = WT + (size_t)(w*32 + 16 + n16)*Cin;   // B column, nt=1
  v8f acc0 = {}, acc1 = {};
  for (int k0 = 0; k0 < Cin; k0 += 32) {
    v16h A  = frag32(xa  + k0, grp);
    v16h B0 = frag32(wb0 + k0, grp);
    v16h B1 = frag32(wb1 + k0, grp);
    acc0 = wmma_f16(A, B0, acc0);
    acc1 = wmma_f16(A, B1, acc1);
  }
  #pragma unroll
  for (int j = 0; j < 8; ++j) {
    int r = j + 8*grp;
    ldsAcc[r*256 + w*32 + n16]      = acc0[j];
    ldsAcc[r*256 + w*32 + 16 + n16] = acc1[j];
  }
  __syncthreads();
  if (t < 16) {
    float m = 0.f;
    for (int c = 0; c < 256; ++c) m += ldsAcc[t*256 + c];
    m *= (1.f/256.f);
    float v = 0.f;
    for (int c = 0; c < 256; ++c) { float d = ldsAcc[t*256 + c] - m; v += d*d; }
    s_mean[t] = m; s_rstd[t] = rsqrtf(v*(1.f/256.f) + EPSF);
  }
  __syncthreads();
  int r = t >> 4, cb = (t & 15)*16;
  float m = s_mean[r], rs = s_rstd[r];
  #pragma unroll
  for (int j = 0; j < 16; ++j) {
    int cc = cb + j;
    float yv = (ldsAcc[r*256 + cc] - m)*rs*gamma[cc] + beta[cc];
    Y[(size_t)(rowbase + r)*EE + cc] = (h16)yv;
  }
}

// ---- stage 3: flash attention, double-buffered K/V pipeline ----------------
// block: 128 thr = 4 waves; one (b, slab, head, 64-query tile) per block.
// Per iteration: dump staged regs -> LDS[cur]; barrier; issue next chunk's
// global loads; compute 16 WMMAs + online softmax on LDS[cur].  The loadcnt
// wait for the staged regs lands after the WMMA block -> latency hidden.
__global__ __launch_bounds__(128)
void attn(const h16* __restrict__ Qh, const h16* __restrict__ Kh, const h16* __restrict__ Vh,
          const float* __restrict__ topo, const float* __restrict__ lam_p,
          h16* __restrict__ Obuf) {
  __shared__ h16 ldsQ[64*72];
  __shared__ h16 ldsK[2][64*72];
  __shared__ h16 ldsVt[2][64*72];                 // transposed: [hd][key]
  __shared__ h16 ldsP[4][16*72];
  int t = threadIdx.x, w = t >> 5, lane = t & 31;
  int n16 = lane & 15, grp = lane >> 4;
  int tile = blockIdx.x, head = blockIdx.y, bk = blockIdx.z;
  int b = bk >> 3, kk = bk & 7;
  int zloc = tile / 9, hwb = (tile % 9)*64;
  int zq = kk*4 + zloc;                           // D0=128, slab=16 -> 4 z per slab
  float lam = lam_p[0];

  // candidate slabs (uniform per block)
  int cand0 = (kk > 0) ? kk - 1 : kk;
  int nlo   = (kk > 0) ? 1 : 0;
  int L     = 1 + nlo + ((kk < KK_-1) ? 1 : 0);
  int total = L*9;
  const h16* KVbase = Kh;                         // silence unused warnings path
  (void)KVbase;

  // per-thread staging slots: e = t + i*128 -> r = e>>3, go = (e&7)*8
  int srow[4], sgo[4];
  #pragma unroll
  for (int i = 0; i < 4; ++i) { int e = t + i*128; srow[i] = e >> 3; sgo[i] = (e & 7)*8; }

  // stage Q tile (64 rows x HD=64) into LDS
  #pragma unroll
  for (int i = 0; i < 4; ++i)
    *(v8h*)&ldsQ[srow[i]*72 + sgo[i]] =
        *(const v8h*)(Qh + ((size_t)(b*DZ + zq)*HW_ + hwb + srow[i])*EE + head*HD_ + sgo[i]);
  __syncthreads();
  v16h QA0, QA1;
  { const h16* base = &ldsQ[(w*16 + n16)*72];
    QA0 = frag32(base, grp);
    QA1 = frag32(base + 32, grp); }

  v8f acc[4]; acc[0] = (v8f){}; acc[1] = (v8f){}; acc[2] = (v8f){}; acc[3] = (v8f){};
  float mrow[8], lrow[8];
  #pragma unroll
  for (int j = 0; j < 8; ++j) { mrow[j] = -1e30f; lrow[j] = 0.f; }
  const int q_hw_base = hwb + w*16;
  const float scale = 0.125f;                     // 1/sqrt(HD)

  // prologue: fetch chunk 0
  v8h rk[4], rv[4];
  {
    int c0 = cand0;
    const h16* Kc = Kh + ((size_t)(b*KK_ + c0)*HW_)*EE + head*HD_;
    const h16* Vc = Vh + ((size_t)(b*KK_ + c0)*HW_)*EE + head*HD_;
    #pragma unroll
    for (int i = 0; i < 4; ++i) {
      rk[i] = *(const v8h*)(Kc + (size_t)srow[i]*EE + sgo[i]);
      rv[i] = *(const v8h*)(Vc + (size_t)srow[i]*EE + sgo[i]);
    }
  }

  for (int it = 0; it < total; ++it) {
    int cur = it & 1;
    int ci = it / 9, chunk = it - ci*9;
    int c  = cand0 + ci;
    int kb = chunk*64;
    // dump staged chunk into LDS[cur] (K row-major; V transposed)
    #pragma unroll
    for (int i = 0; i < 4; ++i) {
      *(v8h*)&ldsK[cur][srow[i]*72 + sgo[i]] = rk[i];
      #pragma unroll
      for (int q2 = 0; q2 < 8; ++q2) ldsVt[cur][(sgo[i] + q2)*72 + srow[i]] = rv[i][q2];
    }
    __syncthreads();
    // issue next chunk's loads (latency hidden behind the WMMA block below)
    if (it + 1 < total) {
      int it2 = it + 1;
      int ci2 = it2 / 9, chunk2 = it2 - ci2*9;
      int c2 = cand0 + ci2;
      int kb2 = chunk2*64;
      const h16* Kc = Kh + ((size_t)(b*KK_ + c2)*HW_)*EE + head*HD_;
      const h16* Vc = Vh + ((size_t)(b*KK_ + c2)*HW_)*EE + head*HD_;
      #pragma unroll
      for (int i = 0; i < 4; ++i) {
        rk[i] = *(const v8h*)(Kc + (size_t)(kb2 + srow[i])*EE + sgo[i]);
        rv[i] = *(const v8h*)(Vc + (size_t)(kb2 + srow[i])*EE + sgo[i]);
      }
    }
    float sb = (c == kk) ? 0.f : -0.5f;           // BETA_ADJ
    const float* tpc = topo + (b*KK_ + c)*HW_;
    // logits S = Q K^T (16 queries x 64 keys) in four 16x16 C tiles
    v8f s[4];
    #pragma unroll
    for (int nt = 0; nt < 4; ++nt) {
      const h16* kr = &ldsK[cur][(nt*16 + n16)*72];
      v8f sv = {};
      sv = wmma_f16(QA0, frag32(kr, grp), sv);
      sv = wmma_f16(QA1, frag32(kr + 32, grp), sv);
      int key_hw = kb + nt*16 + n16;
      float tv = lam * tpc[key_hw];
      #pragma unroll
      for (int j = 0; j < 8; ++j) {
        float val = sv[j]*scale + sb;
        int q_hw = q_hw_base + j + 8*grp;
        sv[j] = (key_hw == q_hw) ? (val + tv) : val;
      }
      s[nt] = sv;
    }
    // online softmax update per row
    #pragma unroll
    for (int j = 0; j < 8; ++j) {
      float mt = fmaxf(fmaxf(s[0][j], s[1][j]), fmaxf(s[2][j], s[3][j]));
      mt = redmax16(mt);
      float mnew = fmaxf(mrow[j], mt);
      float f = __expf(mrow[j] - mnew);
      float rs = 0.f;
      #pragma unroll
      for (int nt = 0; nt < 4; ++nt) {
        float p = __expf(s[nt][j] - mnew);
        s[nt][j] = p; rs += p;
      }
      rs = redsum16(rs);
      lrow[j] = lrow[j]*f + rs;
      mrow[j] = mnew;
      #pragma unroll
      for (int nt = 0; nt < 4; ++nt) acc[nt][j] *= f;
    }
    // route P (C layout) -> LDS -> A layout
    #pragma unroll
    for (int j = 0; j < 8; ++j) {
      int r = j + 8*grp;
      #pragma unroll
      for (int nt = 0; nt < 4; ++nt)
        ldsP[w][r*72 + nt*16 + n16] = (h16)s[nt][j];
    }
    asm volatile("s_wait_dscnt 0" ::: "memory");  // wave-local LDS RAW
    v16h PA0, PA1;
    { const h16* pr = &ldsP[w][n16*72];
      PA0 = frag32(pr, grp);
      PA1 = frag32(pr + 32, grp); }
    #pragma unroll
    for (int nt = 0; nt < 4; ++nt) {
      const h16* vr = &ldsVt[cur][(nt*16 + n16)*72];
      acc[nt] = wmma_f16(PA0, frag32(vr, grp), acc[nt]);
      acc[nt] = wmma_f16(PA1, frag32(vr + 32, grp), acc[nt]);
    }
  }
  // epilogue: O = acc / l -> (B, Dz, HW, E) f16
  #pragma unroll
  for (int j = 0; j < 8; ++j) {
    int r = j + 8*grp;
    int q_hw = hwb + w*16 + r;
    float inv = 1.f / lrow[j];
    size_t orow = ((size_t)(b*DZ + zq)*HW_ + q_hw)*EE + head*HD_;
    #pragma unroll
    for (int nt = 0; nt < 4; ++nt)
      Obuf[orow + nt*16 + n16] = (h16)(acc[nt][j]*inv);
  }
}

// ---- stage 4: out = F3 + O @ Wp (WMMA, fused residual + NCHW scatter) ------
__global__ __launch_bounds__(256)
void proj_out(const h16* __restrict__ Obuf, const h16* __restrict__ WpT,
              const float* __restrict__ F3, float* __restrict__ out) {
  int t = threadIdx.x, w = t >> 5, lane = t & 31;
  int n16 = lane & 15, grp = lane >> 4;
  int rowbase = blockIdx.x * 16;
  const h16* oa = Obuf + (size_t)(rowbase + n16)*EE;
  const h16* wb = WpT  + (size_t)(w*16 + n16)*EE;
  v8f acc = {};
  #pragma unroll
  for (int k0 = 0; k0 < EE; k0 += 32) {
    v16h A  = frag32(oa + k0, grp);
    v16h Bt = frag32(wb + k0, grp);
    acc = wmma_f16(A, Bt, acc);
  }
  #pragma unroll
  for (int j = 0; j < 8; ++j) {
    int r = j + 8*grp;
    int row = rowbase + r;
    int b = row / (DZ*HW_), rem = row % (DZ*HW_);
    int z = rem / HW_, hw = rem % HW_;
    int c = w*16 + n16;
    size_t oi = ((size_t)(b*C3_ + c)*DZ + z)*HW_ + hw;
    out[oi] = F3[oi] + acc[j];
  }
}

// ---- host-side orchestration ----------------------------------------------
extern "C" void kernel_launch(void* const* d_in, const int* in_sizes, int n_in,
                              void* d_out, int out_size, void* d_ws, size_t ws_size,
                              hipStream_t stream) {
  const float* F3     = (const float*)d_in[0];
  const float* F2     = (const float*)d_in[1];
  const float* Wq     = (const float*)d_in[2];
  const float* Wk     = (const float*)d_in[3];
  const float* Wv     = (const float*)d_in[4];
  const float* Wp     = (const float*)d_in[5];
  const float* ln_q_g = (const float*)d_in[6];
  const float* ln_q_b = (const float*)d_in[7];
  const float* ln_kv_g= (const float*)d_in[8];
  const float* ln_kv_b= (const float*)d_in[9];
  const float* conv1w = (const float*)d_in[10];
  const float* bn_g   = (const float*)d_in[11];
  const float* bn_b   = (const float*)d_in[12];
  const float* conv2w = (const float*)d_in[13];
  const float* conv2b = (const float*)d_in[14];
  const float* lam    = (const float*)d_in[15];

  char* ws = (char*)d_ws;
  h16*  Xq    = (h16*)(ws + WS_XQ);
  h16*  Xkv   = (h16*)(ws + WS_XKV);
  h16*  WqT   = (h16*)(ws + WS_WQT);
  h16*  WkT   = (h16*)(ws + WS_WKT);
  h16*  WvT   = (h16*)(ws + WS_WVT);
  h16*  WpT   = (h16*)(ws + WS_WPT);
  h16*  Qb    = (h16*)(ws + WS_Q);
  h16*  Kb    = (h16*)(ws + WS_K);
  h16*  Vb    = (h16*)(ws + WS_V);
  h16*  Ob    = (h16*)(ws + WS_O);
  float* hbuf = (float*)(ws + WS_H);
  float* pbuf = (float*)(ws + WS_P);
  float* topo = (float*)(ws + WS_T);

  // stage 0: transpose-convert weights, pack activations to f16 row-major
  cvt_transpose_f16<<<128, 256, 0, stream>>>(Wq, WqT, 128, 256);
  cvt_transpose_f16<<< 96, 256, 0, stream>>>(Wk, WkT,  96, 256);
  cvt_transpose_f16<<< 96, 256, 0, stream>>>(Wv, WvT,  96, 256);
  cvt_transpose_f16<<<128, 256, 0, stream>>>(Wp, WpT, 256, 128);
  pack_xq <<<18432, 256, 0, stream>>>(F3, Xq);
  pack_xkv<<< 3456, 256, 0, stream>>>(F2, Xkv);
  // stage 1: topo branch
  topo_conv1<<<2304, 256, 0, stream>>>(F2, conv1w, bn_g, bn_b, hbuf);
  topo_conv2<<<  36, 256, 0, stream>>>(hbuf, conv2w, conv2b, pbuf);
  topo_nei  <<<  36, 256, 0, stream>>>(pbuf, topo);
  // stage 2: projections + LN (WMMA)
  gemm_ln<<<2304, 256, 0, stream>>>(Xq,  WqT, ln_q_g,  ln_q_b,  Qb, 128);
  gemm_ln<<< 576, 256, 0, stream>>>(Xkv, WkT, ln_kv_g, ln_kv_b, Kb, 96);
  gemm_ln<<< 576, 256, 0, stream>>>(Xkv, WvT, ln_kv_g, ln_kv_b, Vb, 96);
  // stage 3: flash attention (WMMA, double-buffered)
  attn<<<dim3(36, 4, 16), 128, 0, stream>>>(Qb, Kb, Vb, topo, lam, Ob);
  // stage 4: output projection + residual (WMMA)
  proj_out<<<2304, 256, 0, stream>>>(Ob, WpT, F3, (float*)d_out);
}
